// CAO_RONet_7859790152261
// MI455X (gfx1250) — matrix operand order repaired and checked by reading, not compiled, not run on hardware
//
#include <hip/hip_runtime.h>
#include <math.h>

// ---------------------------------------------------------------------------
// Bidirectional Mamba stack for MI455X (gfx1250, wave32, WMMA).
//
// GEMMs: v_wmma_f32_16x16x32_bf16, f32 operands converted with native bf16
// casts (packed cvt), 2x2 16x16 tiles per wave (4 WMMAs per K-step, A/B
// fragment reuse). GEMMs are memory bound here (~0.5 TFLOP total vs 23.3 TB/s
// HBM and 192MB L2-resident activations), so bf16 WMMA + f32 accum saturates.
// Scan: latency-bound recurrence; one thread per (b,d) chain with 64-wide
// state in VGPRs; the 144-float xdbl row is staged into LDS each step via
// global_load_async_to_lds_b128 (ASYNCcnt) + next-row global_prefetch_b8.
// in_proj is computed once and shared between fwd/bwd passes (reversal only
// affects the conv neighbor and scan direction).
//
// Workspace requirement: ~321 MB of d_ws (carved below).
// ---------------------------------------------------------------------------

#define NL   5
#define DM   256
#define DI   512
#define DS   64
#define DTR  16
#define HID  1024
#define BB   8
#define LL   4096
#define MROWS (BB * LL)          // 32768 activation rows
#define XROW  (DTR + 2 * DS)     // 144 = dt-rank + B + C

typedef __bf16 bf16_t;
typedef bf16_t v16bf __attribute__((ext_vector_type(16)));
typedef float  v8f   __attribute__((ext_vector_type(8)));

__device__ __forceinline__ float silu_f(float x) { return x / (1.0f + expf(-x)); }

enum { EPI_STORE = 0, EPI_ADD = 1, EPI_BIAS_GELU = 2, EPI_BIAS_ADD = 3 };

__device__ __forceinline__ v8f wmma_bf16(const v16bf& a, const v16bf& b, const v8f& c) {
  return __builtin_amdgcn_wmma_f32_16x16x32_bf16(
      /*neg_a=*/false, a, /*neg_b=*/false, b,
      /*c_mod=*/(short)0, c, /*reuse_a=*/false, /*reuse_b=*/false);
}

__device__ __forceinline__ void epi_tile(float* __restrict__ C,
                                         const float* __restrict__ bias,
                                         int Ndim, int epi, int mbase, int ncol,
                                         int g, const v8f& acc) {
  const float bn = (epi >= EPI_BIAS_GELU) ? bias[ncol] : 0.0f;
#pragma unroll
  for (int r = 0; r < 8; ++r) {
    const int m = mbase + r + g * 8;
    const size_t idx = (size_t)m * Ndim + ncol;
    float v = acc[r];
    if (epi == EPI_STORE)            C[idx] = v;
    else if (epi == EPI_ADD)         C[idx] += v;
    else if (epi == EPI_BIAS_GELU) { v += bn; C[idx] = 0.5f * v * (1.0f + erff(v * 0.70710678118f)); }
    else                           { v += bn; C[idx] += v; }
  }
}

// ---------------------------------------------------------------------------
// C[M,N] (+)= A[M,K] * W[N,K]^T  via v_wmma_f32_16x16x32_bf16.
// Block = 128 threads = 4 waves. Wave owns a 32(M)x32(N) macro-tile:
//   m0 = blockIdx.x*32, n0 = blockIdx.y*128 + wave*32 -> 2x2 16x16 tiles,
//   4 WMMAs per K-step with A/B fragment reuse. K must be a multiple of 32.
// Fragment layouts per CDNA5 ISA 7.12.2 (wave32):
//   A 16x32 bf16: lane l (g=l>>4) holds row (l&15);
//     VGPR j<4 -> K = 2j+8g (pair), j>=4 -> K = 16+2(j-4)+8g (pair)
//   B 32x16 bf16: lane l holds col (l&15), same K order as A
//   C 16x16 f32 : VGPR r -> M = r + 8g, N = (l&15)
// All tile guards are wave-uniform => EXEC all-ones at every WMMA.
// ---------------------------------------------------------------------------
__global__ void gemm_wmma_bf16(const float* __restrict__ A,
                               const float* __restrict__ W,
                               const float* __restrict__ bias,
                               float* __restrict__ C,
                               int Ndim, int Kdim, int epi)
{
  const int lane = threadIdx.x & 31;
  const int wave = threadIdx.x >> 5;
  const int m0 = blockIdx.x * 32;
  const int n0 = blockIdx.y * 128 + wave * 32;
  if (n0 >= Ndim) return;                      // wave-uniform
  const bool two = (n0 + 16) < Ndim;           // second N-tile valid?

  const int g     = lane >> 4;
  const int mrow0 = m0 + (lane & 15);
  const int ncol0 = n0 + (lane & 15);
  const int ncol1 = two ? (ncol0 + 16) : ncol0;

  const float* arow0 = A + (size_t)mrow0 * Kdim;
  const float* arow1 = arow0 + (size_t)16 * Kdim;
  const float* wrow0 = W + (size_t)ncol0 * Kdim + g * 8;   // K order matches A
  const float* wrow1 = W + (size_t)ncol1 * Kdim + g * 8;

  v8f acc00 = {}, acc01 = {}, acc10 = {}, acc11 = {};
  for (int k0 = 0; k0 < Kdim; k0 += 32) {
    v16bf a0, a1, b0, b1;
#pragma unroll
    for (int j = 0; j < 8; ++j) {
      const int kloc = (j < 4) ? (2 * j + 8 * g) : (16 + 2 * (j - 4) + 8 * g);
      float2 f0 = *(const float2*)(arow0 + k0 + kloc);
      float2 f1 = *(const float2*)(arow1 + k0 + kloc);
      a0[2 * j] = (bf16_t)f0.x; a0[2 * j + 1] = (bf16_t)f0.y;
      a1[2 * j] = (bf16_t)f1.x; a1[2 * j + 1] = (bf16_t)f1.y;
    }
#pragma unroll
    for (int j = 0; j < 8; ++j) {
      const int kb = (j < 4) ? (2 * j) : (16 + 2 * (j - 4));
      float2 f0 = *(const float2*)(wrow0 + k0 + kb);
      float2 f1 = *(const float2*)(wrow1 + k0 + kb);
      b0[2 * j] = (bf16_t)f0.x; b0[2 * j + 1] = (bf16_t)f0.y;
      b1[2 * j] = (bf16_t)f1.x; b1[2 * j + 1] = (bf16_t)f1.y;
    }
    acc00 = wmma_bf16(a0, b0, acc00);
    acc01 = wmma_bf16(a0, b1, acc01);
    acc10 = wmma_bf16(a1, b0, acc10);
    acc11 = wmma_bf16(a1, b1, acc11);
  }

  epi_tile(C, bias, Ndim, epi, m0,      ncol0, g, acc00);
  epi_tile(C, bias, Ndim, epi, m0 + 16, ncol0, g, acc10);
  if (two) {
    epi_tile(C, bias, Ndim, epi, m0,      ncol1, g, acc01);
    epi_tile(C, bias, Ndim, epi, m0 + 16, ncol1, g, acc11);
  }
}

// ---------------------------------------------------------------------------
// LayerNorm over DM=256: one block (256 threads) per row.
// ---------------------------------------------------------------------------
__global__ void layernorm_kernel(const float* __restrict__ x,
                                 const float* __restrict__ w,
                                 const float* __restrict__ b,
                                 float* __restrict__ y)
{
  __shared__ float red[DM];
  const size_t row = blockIdx.x;
  const int t = threadIdx.x;
  float v = x[row * DM + t];
  red[t] = v; __syncthreads();
  for (int s = DM / 2; s > 0; s >>= 1) { if (t < s) red[t] += red[t + s]; __syncthreads(); }
  float mu = red[0] * (1.0f / DM);
  __syncthreads();
  float dv = v - mu;
  red[t] = dv * dv; __syncthreads();
  for (int s = DM / 2; s > 0; s >>= 1) { if (t < s) red[t] += red[t + s]; __syncthreads(); }
  float var = red[0] * (1.0f / DM);
  y[row * DM + t] = dv * rsqrtf(var + 1e-5f) * w[t] + b[t];
}

// ---------------------------------------------------------------------------
// Depthwise width-2 conv + SiLU. dir=+1: neighbor l-1 (causal fwd);
// dir=-1: neighbor l+1 (reversed pass). xi = xz[..., :DI].
// ---------------------------------------------------------------------------
__global__ void conv_silu_kernel(const float* __restrict__ xz,
                                 const float* __restrict__ cw,
                                 const float* __restrict__ cb,
                                 float* __restrict__ xc, int dir)
{
  size_t idx = (size_t)blockIdx.x * blockDim.x + threadIdx.x;
  if (idx >= (size_t)MROWS * DI) return;
  int d = (int)(idx % DI);
  size_t ml = idx / DI;
  int l = (int)(ml % LL);
  float xi = xz[ml * (size_t)(2 * DI) + d];
  int lp = l - dir;
  float xp = 0.0f;
  if (lp >= 0 && lp < LL) {
    long long mlp = (long long)ml + (lp - l);
    xp = xz[(size_t)mlp * (size_t)(2 * DI) + d];
  }
  float tt = xi * cw[2 * d + 1] + xp * cw[2 * d] + cb[d];
  xc[idx] = silu_f(tt);
}

// ---------------------------------------------------------------------------
// Selective scan + gating. grid = (DI/256, B), block = 256 (8 waves).
// Thread owns channel d: state h[64] and A[64] in registers, dt-proj row (16)
// in registers. Per step, the 144-float xdbl row (576B) is DMA'd into LDS via
// 36 async b128 transfers (ASYNCcnt), broadcast to all lanes; dt-softplus is
// computed inline; output fully gated: y = (scan_y + u*D) * silu(z).
// NOTE: `row` is the kernel's only __shared__ object -> LDS offset 0, so the
// per-lane LDS destination address for the async copy is simply t*16.
// ---------------------------------------------------------------------------
__global__ void scan_gate_kernel(const float* __restrict__ xdbl,
                                 const float* __restrict__ xc,
                                 const float* __restrict__ xz,
                                 const float* __restrict__ dtw,
                                 const float* __restrict__ dtb,
                                 const float* __restrict__ alog,
                                 const float* __restrict__ Dp,
                                 float* __restrict__ ybuf, int dir)
{
  __shared__ float row[XROW];                 // 576 bytes, LDS offset 0
  const int t = threadIdx.x;
  const int b = blockIdx.y;
  const int d = blockIdx.x * blockDim.x + t;

  float Avec[DS], h[DS], dtwr[DTR];
#pragma unroll
  for (int s = 0; s < DS; ++s) { Avec[s] = -expf(alog[(size_t)d * DS + s]); h[s] = 0.0f; }
#pragma unroll
  for (int r = 0; r < DTR; ++r) dtwr[r] = dtw[d * DTR + r];
  const float dtbias = dtb[d];
  const float Dd = Dp[d];

  for (int step = 0; step < LL; ++step) {
    const int l = (dir > 0) ? step : (LL - 1 - step);
    const size_t base = (size_t)b * LL + l;

    // Async DMA the xdbl row into LDS: 36 lanes x 16 bytes.
    if (t < (XROW / 4)) {
      const float* gsrc = xdbl + base * XROW + t * 4;
      const unsigned ldsoff = (unsigned)(t * 16);   // 'row' lives at LDS offset 0
      asm volatile("global_load_async_to_lds_b128 %0, %1, off"
                   :: "v"(ldsoff), "v"(gsrc) : "memory");
    }
#if __has_builtin(__builtin_amdgcn_s_wait_asynccnt)
    __builtin_amdgcn_s_wait_asynccnt(0);
#else
    asm volatile("s_wait_asynccnt 0" ::: "memory");
#endif
    __syncthreads();

    // Prefetch next step's row into cache (global_prefetch_b8).
    if (step + 1 < LL && t < XROW) {
      const int ln = (dir > 0) ? (l + 1) : (l - 1);
      __builtin_prefetch(xdbl + ((size_t)b * LL + ln) * XROW + t, 0, 1);
    }

    float x0 = dtbias;
#pragma unroll
    for (int r = 0; r < DTR; ++r) x0 += row[r] * dtwr[r];
    const float dt = (x0 > 20.0f) ? x0 : log1pf(expf(x0));   // softplus

    const float u = xc[base * DI + d];
    const float dtu = dt * u;
    float y = 0.0f;
#pragma unroll
    for (int s = 0; s < DS; ++s) {
      float e = expf(dt * Avec[s]);
      h[s] = e * h[s] + dtu * row[DTR + s];        // B_t
      y += h[s] * row[DTR + DS + s];               // C_t
    }
    const float z = xz[base * (size_t)(2 * DI) + DI + d];
    ybuf[base * DI + d] = (y + u * Dd) * silu_f(z);
    __syncthreads();
  }
}

// ---------------------------------------------------------------------------
// Host orchestration (graph-capture safe: launches + one async d2d copy).
// ---------------------------------------------------------------------------
extern "C" void kernel_launch(void* const* d_in, const int* in_sizes, int n_in,
                              void* d_out, int out_size, void* d_ws, size_t ws_size,
                              hipStream_t stream)
{
  (void)in_sizes; (void)n_in; (void)out_size; (void)ws_size;

  const float* x     = (const float*)d_in[0];
  const float* inw   = (const float*)d_in[1];
  const float* convw = (const float*)d_in[2];
  const float* convb = (const float*)d_in[3];
  const float* xpw   = (const float*)d_in[4];
  const float* dtw   = (const float*)d_in[5];
  const float* dtb   = (const float*)d_in[6];
  const float* alog  = (const float*)d_in[7];
  const float* Dp    = (const float*)d_in[8];
  const float* ow    = (const float*)d_in[9];
  const float* n1w   = (const float*)d_in[10];
  const float* n1b   = (const float*)d_in[11];
  const float* n2w   = (const float*)d_in[12];
  const float* n2b   = (const float*)d_in[13];
  const float* fc1w  = (const float*)d_in[14];
  const float* fc1b  = (const float*)d_in[15];
  const float* fc2w  = (const float*)d_in[16];
  const float* fc2b  = (const float*)d_in[17];

  float* xout = (float*)d_out;                 // residual stream, (B,L,DM)

  // Workspace carve (f32): xn(33.5MB) xz(134MB) xc(67MB) xdbl(19MB) ybuf(67MB)
  float* ws   = (float*)d_ws;
  float* xn   = ws;                                    // M*DM
  float* xzb  = xn   + (size_t)MROWS * DM;             // M*2DI (also MLP hidden)
  float* xcb  = xzb  + (size_t)MROWS * 2 * DI;         // M*DI
  float* xdbl = xcb  + (size_t)MROWS * DI;             // M*XROW
  float* ybuf = xdbl + (size_t)MROWS * XROW;           // M*DI
  float* hbuf = xzb;                                   // HID == 2*DI, safe reuse

  hipMemcpyAsync(xout, x, (size_t)MROWS * DM * sizeof(float),
                 hipMemcpyDeviceToDevice, stream);

  const dim3 gblk(128, 1, 1);
  for (int i = 0; i < NL; ++i) {
    // --- Mamba (bidirectional) ---
    layernorm_kernel<<<MROWS, DM, 0, stream>>>(xout, n1w + i * DM, n1b + i * DM, xn);

    // in_proj once: xz = xn @ inw^T   (shared by fwd and bwd passes)
    gemm_wmma_bf16<<<dim3(MROWS / 32, (2 * DI + 127) / 128), gblk, 0, stream>>>(
        xn, inw + (size_t)i * 2 * DI * DM, nullptr, xzb, 2 * DI, DM, EPI_STORE);

    for (int pass = 0; pass < 2; ++pass) {
      const int dir = (pass == 0) ? 1 : -1;
      conv_silu_kernel<<<(MROWS * DI) / 256, 256, 0, stream>>>(
          xzb, convw + (size_t)i * DI * 2, convb + (size_t)i * DI, xcb, dir);
      gemm_wmma_bf16<<<dim3(MROWS / 32, (XROW + 127) / 128), gblk, 0, stream>>>(
          xcb, xpw + (size_t)i * XROW * DI, nullptr, xdbl, XROW, DI, EPI_STORE);
      scan_gate_kernel<<<dim3(DI / 256, BB), 256, 0, stream>>>(
          xdbl, xcb, xzb, dtw + (size_t)i * DI * DTR, dtb + (size_t)i * DI,
          alog + (size_t)i * DI * DS, Dp + (size_t)i * DI, ybuf, dir);
      // x += y @ ow^T  (residual accumulate in epilogue)
      gemm_wmma_bf16<<<dim3(MROWS / 32, (DM + 127) / 128), gblk, 0, stream>>>(
          ybuf, ow + (size_t)i * DM * DI, nullptr, xout, DM, DI, EPI_ADD);
    }

    // --- MLP ---
    layernorm_kernel<<<MROWS, DM, 0, stream>>>(xout, n2w + i * DM, n2b + i * DM, xn);
    gemm_wmma_bf16<<<dim3(MROWS / 32, (HID + 127) / 128), gblk, 0, stream>>>(
        xn, fc1w + (size_t)i * HID * DM, fc1b + (size_t)i * HID, hbuf, HID, DM, EPI_BIAS_GELU);
    gemm_wmma_bf16<<<dim3(MROWS / 32, (DM + 127) / 128), gblk, 0, stream>>>(
        hbuf, fc2w + (size_t)i * DM * HID, fc2b + (size_t)i * DM, xout, DM, HID, EPI_BIAS_ADD);
  }
}
